// MSA_1486058684965
// MI455X (gfx1250) — compile-verified, matrix operand order checked by reading, not run on hardware
//
#include <hip/hip_runtime.h>
#include <hip/hip_bf16.h>

typedef __attribute__((ext_vector_type(16))) __bf16 v16bf;
typedef __attribute__((ext_vector_type(8)))  __bf16 v8bf;
typedef __attribute__((ext_vector_type(8)))  float  v8f;

#define DIMD   768
#define NHEAD  12
#define HD     64
#define BATCH  4
#define SEQ    2048
#define BN     (BATCH * SEQ)     // 8192
#define QKVD   (3 * DIMD)        // 2304

// SCALE * log2(e) folded into Q so softmax can use native v_exp_f32 (base 2)
#define QSCALE_LOG2E 0.18033688011112042f   // (1/8) * 1.4426950408889634

// ---------------------------------------------------------------------------
// Fragment loaders (CDNA5 wave32 WMMA layouts, CDNA5 ISA §7.12.2)
// A-matrix 16x32 bf16: lane L holds row M=L%16; element i maps to
//   K = (i/8)*16 + (L/16)*8 + (i%8)   -> two contiguous 8-element (16B) loads.
// B (as N x K, row-major) uses the identical mapping with N = L%16.
// ---------------------------------------------------------------------------
__device__ __forceinline__ v16bf load_frag_bf16(const __bf16* __restrict__ base,
                                                int row0, int k0, int ld) {
  const int lane = threadIdx.x & 31;
  const __bf16* p = base + (size_t)(row0 + (lane & 15)) * ld + k0 + ((lane >> 4) << 3);
  v8bf lo = *(const v8bf*)p;          // K = half*8 + 0..7
  v8bf hi = *(const v8bf*)(p + 16);   // K = 16 + half*8 + 0..7
  v16bf r;
#pragma unroll
  for (int i = 0; i < 8; ++i) { r[i] = lo[i]; r[i + 8] = hi[i]; }
  return r;
}

// A-fragment read of a wave-private 16x32 bf16 tile in LDS (row stride 32)
__device__ __forceinline__ v16bf load_frag_lds(const __bf16* p) {
  const int lane = threadIdx.x & 31;
  const __bf16* q = p + (lane & 15) * 32 + ((lane >> 4) << 3);
  v8bf lo = *(const v8bf*)q;
  v8bf hi = *(const v8bf*)(q + 16);
  v16bf r;
#pragma unroll
  for (int i = 0; i < 8; ++i) { r[i] = lo[i]; r[i + 8] = hi[i]; }
  return r;
}

__device__ __forceinline__ v8f wmma_bf16(v16bf a, v16bf b, v8f c) {
  return __builtin_amdgcn_wmma_f32_16x16x32_bf16(false, a, false, b,
                                                 (short)0, c, false, false);
}

// ---------------------------------------------------------------------------
// fp32 -> bf16 conversion (grid-stride)
// ---------------------------------------------------------------------------
__global__ void cvt_bf16_kernel(const float* __restrict__ src,
                                __bf16* __restrict__ dst, int n) {
  for (int i = blockIdx.x * blockDim.x + threadIdx.x; i < n;
       i += gridDim.x * blockDim.x)
    dst[i] = (__bf16)src[i];
}

// ---------------------------------------------------------------------------
// Ping-pong GEMM body: wave = 32x64 tile, 8 WMMA / 12 b128-loads per 32-wide
// k-step, two fragment sets alternating roles (no rotation copies). The k
// loop is FULLY UNROLLED so every k offset folds into the 24-bit instruction
// offset (max 1536B here): no address VALU in steady state, no WMMA->VALU
// WAR-hazard nops.
// ---------------------------------------------------------------------------
#define GEMM_LOAD_SET(AMAT, BMAT, A0, A1, BARR, KOFF)                        \
  A0 = load_frag_bf16(AMAT, m0,      (KOFF), DIMD);                          \
  A1 = load_frag_bf16(AMAT, m0 + 16, (KOFF), DIMD);                          \
  _Pragma("unroll")                                                          \
  for (int j = 0; j < 4; ++j)                                                \
    BARR[j] = load_frag_bf16(BMAT, n0 + j * 16, (KOFF), DIMD);

#define GEMM_COMPUTE_SET(A0, A1, BARR)                                       \
  _Pragma("unroll")                                                          \
  for (int j = 0; j < 4; ++j) {                                              \
    cc[0][j] = wmma_bf16(A0, BARR[j], cc[0][j]);                             \
    cc[1][j] = wmma_bf16(A1, BARR[j], cc[1][j]);                             \
  }

#define GEMM_MAIN_LOOP(AMAT, BMAT)                                           \
  v16bf aA0, aA1, bA[4];                                                     \
  v16bf aB0, aB1, bB[4];                                                     \
  GEMM_LOAD_SET(AMAT, BMAT, aA0, aA1, bA, 0)                                 \
  _Pragma("unroll")                                                          \
  for (int k = 0; k < DIMD; k += 64) {                                       \
    /* half-step 0: compute set A, prefetch set B at k+32 */                 \
    GEMM_LOAD_SET(AMAT, BMAT, aB0, aB1, bB, k + 32)                          \
    GEMM_COMPUTE_SET(aA0, aA1, bA)                                           \
    /* half-step 1: compute set B, prefetch set A at k+64 (DCE'd on tail) */ \
    if (k + 64 < DIMD) {                                                     \
      GEMM_LOAD_SET(AMAT, BMAT, aA0, aA1, bA, k + 64)                        \
    }                                                                        \
    GEMM_COMPUTE_SET(aB0, aB1, bB)                                           \
  }

// ---------------------------------------------------------------------------
// QKV GEMM: [8192,768]bf16 @ w_qkv^T bf16 -> q (pre-scaled by SCALE*log2e)
// and k as [B,H,N,64] bf16, v transposed as [B,H,64,N] bf16.
// ---------------------------------------------------------------------------
__global__ __launch_bounds__(256) void qkv_kernel(
    const __bf16* __restrict__ xb, const __bf16* __restrict__ wb,
    __bf16* __restrict__ qg, __bf16* __restrict__ kg, __bf16* __restrict__ vtg) {
  const int lane = threadIdx.x & 31;
  const int wave = threadIdx.x >> 5;
  const int half = lane >> 4;
  const int coln = lane & 15;
  const int m0 = blockIdx.y * 256 + wave * 32;
  const int n0 = blockIdx.x * 64;                 // one (s, head) per 64 cols

  v8f cc[2][4];
#pragma unroll
  for (int g = 0; g < 2; ++g)
#pragma unroll
    for (int j = 0; j < 4; ++j) cc[g][j] = v8f{};

  GEMM_MAIN_LOOP(xb, wb)

  const int s = n0 / DIMD;              // 0=q 1=k 2=v
  const int h = (n0 % DIMD) / HD;
#pragma unroll
  for (int g = 0; g < 2; ++g) {
#pragma unroll
    for (int j = 0; j < 4; ++j) {
#pragma unroll
      for (int r = 0; r < 8; ++r) {
        const int gm = m0 + g * 16 + r + (half << 3);
        const int bb = gm >> 11, t = gm & 2047;
        const int hd = j * 16 + coln;
        const float v = cc[g][j][r];
        if (s == 0)
          qg[(((size_t)(bb * NHEAD + h)) * SEQ + t) * HD + hd] =
              (__bf16)(v * QSCALE_LOG2E);
        else if (s == 1)
          kg[(((size_t)(bb * NHEAD + h)) * SEQ + t) * HD + hd] = (__bf16)v;
        else
          vtg[(((size_t)(bb * NHEAD + h)) * HD + hd) * SEQ + t] = (__bf16)v;
      }
    }
  }
}

// ---------------------------------------------------------------------------
// One 32-key flash-attention sub-block: uses K set KC, prefetches K set KP
// for key block KBPRE, loads V for KBCUR early (latency hidden under the
// S-WMMAs and softmax VALU work).
// ---------------------------------------------------------------------------
#define ATTN_STEP(KC, KP, KBCUR, KBPRE)                                      \
  {                                                                          \
    v16bf vv[4];                                                             \
    _Pragma("unroll")                                                        \
    for (int j = 0; j < 4; ++j)                                              \
      vv[j] = load_frag_bf16(vth, j * 16, (KBCUR), SEQ);                     \
    KP[0] = load_frag_bf16(kh, (KBPRE),       0, HD);                        \
    KP[1] = load_frag_bf16(kh, (KBPRE),      32, HD);                        \
    KP[2] = load_frag_bf16(kh, (KBPRE) + 16,  0, HD);                        \
    KP[3] = load_frag_bf16(kh, (KBPRE) + 16, 32, HD);                        \
    v8f s0 = {}, s1 = {};                                                    \
    s0 = wmma_bf16(aq0, KC[0], s0);                                          \
    s0 = wmma_bf16(aq1, KC[1], s0);                                          \
    s1 = wmma_bf16(aq0, KC[2], s1);                                          \
    s1 = wmma_bf16(aq1, KC[3], s1);                                          \
    _Pragma("unroll")                                                        \
    for (int r = 0; r < 8; ++r) {                                            \
      float v0 = s0[r], v1 = s1[r];                                          \
      float vm = fmaxf(v0, v1);                                              \
      vm = fmaxf(vm, __shfl_xor(vm, 1, 32));                                 \
      vm = fmaxf(vm, __shfl_xor(vm, 2, 32));                                 \
      vm = fmaxf(vm, __shfl_xor(vm, 4, 32));                                 \
      vm = fmaxf(vm, __shfl_xor(vm, 8, 32));                                 \
      const float mnew = fmaxf(mrow[r], vm);                                 \
      const float p0 = __builtin_amdgcn_exp2f(v0 - mnew);                    \
      const float p1 = __builtin_amdgcn_exp2f(v1 - mnew);                    \
      float rs = p0 + p1;                                                    \
      rs += __shfl_xor(rs, 1, 32);                                           \
      rs += __shfl_xor(rs, 2, 32);                                           \
      rs += __shfl_xor(rs, 4, 32);                                           \
      rs += __shfl_xor(rs, 8, 32);                                           \
      const float alpha = __builtin_amdgcn_exp2f(mrow[r] - mnew);            \
      lrow[r] = lrow[r] * alpha + rs;                                        \
      mrow[r] = mnew;                                                        \
      o0[r] *= alpha; o1[r] *= alpha; o2[r] *= alpha; o3[r] *= alpha;        \
      const int row = r + (half << 3);                                       \
      pw[row * 32 + col]      = (__bf16)p0;                                  \
      pw[row * 32 + 16 + col] = (__bf16)p1;                                  \
    }                                                                        \
    const v16bf ap = load_frag_lds(pw);                                      \
    o0 = wmma_bf16(ap, vv[0], o0);                                           \
    o1 = wmma_bf16(ap, vv[1], o1);                                           \
    o2 = wmma_bf16(ap, vv[2], o2);                                           \
    o3 = wmma_bf16(ap, vv[3], o3);                                           \
  }

// ---------------------------------------------------------------------------
// Fused flash attention. Block = (query_block=128, head, batch); wave = 16
// queries over all 2048 keys, 64 keys per iteration in two ping-pong
// sub-blocks (no fragment rotation copies).
// ---------------------------------------------------------------------------
__global__ __launch_bounds__(256) void attn_kernel(
    const __bf16* __restrict__ qg, const __bf16* __restrict__ kg,
    const __bf16* __restrict__ vtg, __bf16* __restrict__ ao) {
  __shared__ __align__(16) __bf16 smem[8 * 16 * 32];   // 8 KB
  const int lane = threadIdx.x & 31;
  const int wave = threadIdx.x >> 5;
  const int half = lane >> 4;
  const int col  = lane & 15;
  const int h = blockIdx.y, b = blockIdx.z;
  const int q0 = blockIdx.x * 128 + wave * 16;
  const size_t head_off = ((size_t)(b * NHEAD + h)) * SEQ * HD;
  const __bf16* qh  = qg  + head_off;
  const __bf16* kh  = kg  + head_off;
  const __bf16* vth = vtg + head_off;   // [64, 2048]
  __bf16* pw = &smem[wave * 512];       // wave-private 16x32 tile

  const v16bf aq0 = load_frag_bf16(qh, q0,  0, HD);
  const v16bf aq1 = load_frag_bf16(qh, q0, 32, HD);

  v8f o0 = {}, o1 = {}, o2 = {}, o3 = {};
  float mrow[8], lrow[8];
#pragma unroll
  for (int r = 0; r < 8; ++r) { mrow[r] = -3.0e38f; lrow[r] = 0.0f; }

  // prologue: K fragments for key block 0
  v16bf kA[4], kB[4];
  kA[0] = load_frag_bf16(kh,  0,  0, HD);
  kA[1] = load_frag_bf16(kh,  0, 32, HD);
  kA[2] = load_frag_bf16(kh, 16,  0, HD);
  kA[3] = load_frag_bf16(kh, 16, 32, HD);

  for (int kb = 0; kb < SEQ; kb += 64) {
    // keys kb..kb+31 with set A; prefetch set B for kb+32 (always in range)
    ATTN_STEP(kA, kB, kb, kb + 32)
    // keys kb+32..kb+63 with set B; prefetch set A for kb+64 (dummy wrap)
    const int k2 = (kb + 64 < SEQ) ? kb + 64 : 0;
    ATTN_STEP(kB, kA, kb + 32, k2)
  }

#pragma unroll
  for (int r = 0; r < 8; ++r) {
    const float inv = 1.0f / lrow[r];
    const int gq = q0 + r + (half << 3);
    const size_t base = ((size_t)(b * SEQ + gq)) * DIMD + h * HD + col;
    ao[base]      = (__bf16)(o0[r] * inv);
    ao[base + 16] = (__bf16)(o1[r] * inv);
    ao[base + 32] = (__bf16)(o2[r] * inv);
    ao[base + 48] = (__bf16)(o3[r] * inv);
  }
}

// ---------------------------------------------------------------------------
// Output projection: [8192,768]bf16 @ w_proj^T bf16 + bias -> fp32 out.
// Same fully-unrolled ping-pong 32x64 wave tile as qkv_kernel.
// ---------------------------------------------------------------------------
__global__ __launch_bounds__(256) void proj_kernel(
    const __bf16* __restrict__ aob, const __bf16* __restrict__ wpb,
    const float* __restrict__ bias, float* __restrict__ out) {
  const int lane = threadIdx.x & 31;
  const int wave = threadIdx.x >> 5;
  const int half = lane >> 4;
  const int coln = lane & 15;
  const int m0 = blockIdx.y * 256 + wave * 32;
  const int n0 = blockIdx.x * 64;

  v8f cc[2][4];
#pragma unroll
  for (int g = 0; g < 2; ++g)
#pragma unroll
    for (int j = 0; j < 4; ++j) cc[g][j] = v8f{};

  GEMM_MAIN_LOOP(aob, wpb)

#pragma unroll
  for (int j = 0; j < 4; ++j) {
    const int n = n0 + j * 16 + coln;
    const float bv = bias[n];
#pragma unroll
    for (int g = 0; g < 2; ++g) {
#pragma unroll
      for (int r = 0; r < 8; ++r) {
        const int gm = m0 + g * 16 + r + (half << 3);
        out[(size_t)gm * DIMD + n] = cc[g][j][r] + bv;
      }
    }
  }
}

// ---------------------------------------------------------------------------
extern "C" void kernel_launch(void* const* d_in, const int* in_sizes, int n_in,
                              void* d_out, int out_size, void* d_ws, size_t ws_size,
                              hipStream_t stream) {
  const float* x      = (const float*)d_in[0];
  const float* w_qkv  = (const float*)d_in[1];
  const float* w_proj = (const float*)d_in[2];
  const float* b_proj = (const float*)d_in[3];
  float* out = (float*)d_out;

  // workspace layout (bf16 buffers, 256B aligned)
  char* ws = (char*)d_ws;
  size_t off = 0;
  auto alloc = [&](size_t elems) {
    size_t o = off;
    off = (off + elems * sizeof(__bf16) + 255) & ~(size_t)255;
    return (__bf16*)(ws + o);
  };
  __bf16* xb     = alloc((size_t)BN * DIMD);
  __bf16* wqkvb  = alloc((size_t)QKVD * DIMD);
  __bf16* wprojb = alloc((size_t)DIMD * DIMD);
  __bf16* qg     = alloc((size_t)BATCH * NHEAD * SEQ * HD);
  __bf16* kg     = alloc((size_t)BATCH * NHEAD * SEQ * HD);
  __bf16* vtg    = alloc((size_t)BATCH * NHEAD * SEQ * HD);
  __bf16* aob    = alloc((size_t)BN * DIMD);

  // 1) fp32 -> bf16 staging
  cvt_bf16_kernel<<<4096, 256, 0, stream>>>(x, xb, BN * DIMD);
  cvt_bf16_kernel<<<2048, 256, 0, stream>>>(w_qkv, wqkvb, QKVD * DIMD);
  cvt_bf16_kernel<<<1024, 256, 0, stream>>>(w_proj, wprojb, DIMD * DIMD);

  // 2) QKV projection (writes q*scale*log2e, k, v^T per head)
  qkv_kernel<<<dim3(QKVD / 64, BN / 256), 256, 0, stream>>>(xb, wqkvb, qg, kg, vtg);

  // 3) fused flash attention
  attn_kernel<<<dim3(SEQ / 128, NHEAD, BATCH), 256, 0, stream>>>(qg, kg, vtg, aob);

  // 4) output projection + bias (fp32 out)
  proj_kernel<<<dim3(DIMD / 64, BN / 256), 256, 0, stream>>>(aob, wprojb, b_proj, out);
}
